// GatedRecurrentBlock_8615704396122
// MI455X (gfx1250) — matrix operand order repaired and checked by reading, not compiled, not run on hardware
//
#include <hip/hip_runtime.h>
#include <hip/hip_bf16.h>
#include <cstdint>
#include <cstddef>

typedef __attribute__((ext_vector_type(16))) _Float16 v16h;
typedef __attribute__((ext_vector_type(8)))  _Float16 h8;
typedef __attribute__((ext_vector_type(8)))  float    v8f;

constexpr int BATCH = 8192;
constexpr int DIM   = 2048;
constexpr int HID   = 5324;   // reference HIDDEN
constexpr int HIDP  = 5376;   // padded to multiple of 256 (zero-padded -> exact)
constexpr int GKDIM = 4096;   // gate K = 2*DIM

// ---------------------------------------------------------------------------
// Elementwise: h = (x+state)*0.7071 ; also stash f16(state) into concat[:,2048:]
// ---------------------------------------------------------------------------
__global__ __launch_bounds__(256) void prep_kernel(const float* __restrict__ x,
                                                   const float* __restrict__ st,
                                                   float* __restrict__ h,
                                                   _Float16* __restrict__ cat) {
  size_t i = (size_t)blockIdx.x * 256 + threadIdx.x;
  int row = (int)(i >> 11);
  int col = (int)(i & 2047);
  float s  = st[i];
  float hv = (x[i] + s) * 0.7071f;
  h[i] = hv;
  cat[(size_t)row * GKDIM + DIM + col] = (_Float16)s;
}

// ---------------------------------------------------------------------------
// Row RMS-norm (one block per row, 2048 elems, fp32 math), f16 output
// ---------------------------------------------------------------------------
__device__ inline float block_sum(float v) {
#pragma unroll
  for (int o = 16; o > 0; o >>= 1) v += __shfl_xor(v, o, 32);
  __shared__ float wsum[8];
  __shared__ float tot;
  const int lane = threadIdx.x & 31, w = threadIdx.x >> 5;
  if (lane == 0) wsum[w] = v;
  __syncthreads();
  if (w == 0) {
    float u = (lane < 8) ? wsum[lane] : 0.f;
#pragma unroll
    for (int o = 4; o > 0; o >>= 1) u += __shfl_xor(u, o, 32);
    if (lane == 0) tot = u;
  }
  __syncthreads();
  return tot;
}

__global__ __launch_bounds__(256) void rmsnorm_kernel(const float* __restrict__ in,
                                                      const float* __restrict__ g,
                                                      _Float16* __restrict__ out) {
  const size_t base = (size_t)blockIdx.x * DIM;
  float v[8];
  float ss = 0.f;
#pragma unroll
  for (int i = 0; i < 8; ++i) {
    int j = threadIdx.x + i * 256;
    v[i] = in[base + j];
    ss += v[i] * v[i];
  }
  const float total = block_sum(ss);
  const float scale = 0.02209708691f / fmaxf(sqrtf(total), 1e-6f);  // DIM^-0.5 / max(||x||,eps)
#pragma unroll
  for (int i = 0; i < 8; ++i) {
    int j = threadIdx.x + i * 256;
    out[base + j] = (_Float16)(v[i] * scale * g[j]);
  }
}

// ---------------------------------------------------------------------------
// SwiGLU combine in place: t1 = silu(t1) * t2
// ---------------------------------------------------------------------------
__global__ __launch_bounds__(256) void swiglu_kernel(_Float16* __restrict__ t1,
                                                     const _Float16* __restrict__ t2,
                                                     long long n) {
  long long i = (long long)blockIdx.x * 256 + threadIdx.x;
  if (i >= n) return;
  float a = (float)t1[i];
  float b = (float)t2[i];
  float s = a / (1.f + expf(-a));
  t1[i] = (_Float16)(s * b);
}

// ---------------------------------------------------------------------------
// Weight convert fp32 -> f16 with zero padding (dst is dRows x dCols)
// ---------------------------------------------------------------------------
__global__ __launch_bounds__(256) void convert_pad_kernel(const float* __restrict__ src,
                                                          _Float16* __restrict__ dst,
                                                          int sRows, int sCols, int dCols,
                                                          long long total) {
  long long i = (long long)blockIdx.x * 256 + threadIdx.x;
  if (i >= total) return;
  int r = (int)(i / dCols);
  int c = (int)(i % dCols);
  float v = (r < sRows && c < sCols) ? src[(long long)r * sCols + c] : 0.f;
  dst[i] = (_Float16)v;
}

// ---------------------------------------------------------------------------
// Tiled f16 WMMA GEMM: C[M,N] = epilogue( A[M,K] @ W[N,K]^T )
// Block tile 128(M) x 256(N), K step 32, 256 threads = 8 wave32 in a
// 2(M) x 4(N) wave grid; each wave owns a 64x64 strip = 4x4 tiles of 16x16
// via v_wmma_f32_16x16x32_f16 (fp32 accum). 16 WMMAs per 16 ds_load_b128
// => 32 FLOP per LDS byte, matching ~128 B/cycle/SIMD LDS bandwidth.
// EPI: 0 = +bias -> f16            (v projection)
//      1 = +bias +res -> f32       (attention out + residual)
//      2 = plain -> f16            (w1 / w2)
//      3 = +res  -> f32 and f16    (candidate; f16 copy into concat buffer)
//      4 = +bias, z=sigmoid, out = z*cand + (1-z)*state -> f32 (gate)
// ---------------------------------------------------------------------------
constexpr int BM = 128, BN = 256, BK = 32;
constexpr int LDSP = 40;  // LDS row stride in halves: 80B (16B aligned, bank-spread)

template <int EPI>
__global__ __launch_bounds__(256) void gemm_kernel(
    const _Float16* __restrict__ A, const _Float16* __restrict__ W,
    const float* __restrict__ bias, const float* __restrict__ res,
    const float* __restrict__ cand, const float* __restrict__ st,
    float* __restrict__ Cf, _Float16* __restrict__ Ch,
    int M, int N, int K, int ldc, int ldch) {
  (void)M; (void)N;
  __shared__ _Float16 sA[2][BM * LDSP];   // 2 x 10 KB
  __shared__ _Float16 sB[2][BN * LDSP];   // 2 x 20 KB

  const int tid  = threadIdx.x;
  const int lane = tid & 31;
  const int wave = tid >> 5;
  const int wm = wave & 1;   // wave M index (0..1) -> 64 rows each
  const int wn = wave >> 1;  // wave N index (0..3) -> 64 cols each
  const int bm = blockIdx.y * BM;
  const int bn = blockIdx.x * BN;

  // Tile loaders.
  // A tile 128x32 halves: thread t -> row (t&127), 16-half chunk ((t>>7)*16)
  const int arow = tid & 127;
  const int acol = (tid >> 7) << 4;
  // B tile 256x32 halves: thread t -> row t, full 32-half row (4 x b128)
  const _Float16* gA = A + (size_t)(bm + arow) * K + acol;
  const _Float16* gW = W + (size_t)(bn + tid) * K;

  v8f c[4][4] = {};
  const int nk = K / BK;

  // Preload tile 0
  {
    h8 a0 = *(const h8*)gA, a1 = *(const h8*)(gA + 8);
    h8 b0 = *(const h8*)gW, b1 = *(const h8*)(gW + 8);
    h8 b2 = *(const h8*)(gW + 16), b3 = *(const h8*)(gW + 24);
    _Float16* sa = &sA[0][arow * LDSP + acol];
    _Float16* sb = &sB[0][tid * LDSP];
    *(h8*)sa = a0; *(h8*)(sa + 8) = a1;
    *(h8*)sb = b0; *(h8*)(sb + 8) = b1;
    *(h8*)(sb + 16) = b2; *(h8*)(sb + 24) = b3;
  }
  __syncthreads();

  int buf = 0;
  for (int kt = 0; kt < nk; ++kt) {
    h8 na0{}, na1{}, nb0{}, nb1{}, nb2{}, nb3{};
    const bool hasNext = (kt + 1) < nk;
    if (hasNext) {  // register prefetch of next K tile while computing
      const _Float16* pA = gA + (size_t)(kt + 1) * BK;
      const _Float16* pW = gW + (size_t)(kt + 1) * BK;
      na0 = *(const h8*)pA;        na1 = *(const h8*)(pA + 8);
      nb0 = *(const h8*)pW;        nb1 = *(const h8*)(pW + 8);
      nb2 = *(const h8*)(pW + 16); nb3 = *(const h8*)(pW + 24);
    }

    // Fragment loads per documented WMMA VGPR layouts.
    const _Float16* pa = sA[buf];
    const _Float16* pb = sB[buf];
    const int koffA = (lane >> 4) * 8;   // A: lanes 0-15 K{0..7,16..23}; 16-31 K{8..15,24..31}
    const int koffB = (lane >> 4) * 8;   // B: lanes 0-15 K{0..7,16..23}; 16-31 K{8..15,24..31}
    const int rA = wm * 64 + (lane & 15);
    const int rB = wn * 64 + (lane & 15);
    v16h af[4], bf[4];
#pragma unroll
    for (int tm = 0; tm < 4; ++tm) {
      const _Float16* p = pa + (rA + tm * 16) * LDSP + koffA;
      h8 lo = *(const h8*)p, hi = *(const h8*)(p + 16);
      af[tm] = __builtin_shufflevector(lo, hi, 0,1,2,3,4,5,6,7,8,9,10,11,12,13,14,15);
    }
#pragma unroll
    for (int tn = 0; tn < 4; ++tn) {
      const _Float16* p = pb + (rB + tn * 16) * LDSP + koffB;
      h8 lo = *(const h8*)p, hi = *(const h8*)(p + 16);
      bf[tn] = __builtin_shufflevector(lo, hi, 0,1,2,3,4,5,6,7,8,9,10,11,12,13,14,15);
    }
#pragma unroll
    for (int tm = 0; tm < 4; ++tm)
#pragma unroll
      for (int tn = 0; tn < 4; ++tn)
        c[tm][tn] = __builtin_amdgcn_wmma_f32_16x16x32_f16(
            false, af[tm], false, bf[tn], (short)0, c[tm][tn], false, false);

    __syncthreads();
    if (hasNext) {
      _Float16* sa = &sA[buf ^ 1][arow * LDSP + acol];
      _Float16* sb = &sB[buf ^ 1][tid * LDSP];
      *(h8*)sa = na0; *(h8*)(sa + 8) = na1;
      *(h8*)sb = nb0; *(h8*)(sb + 8) = nb1;
      *(h8*)(sb + 16) = nb2; *(h8*)(sb + 24) = nb3;
      buf ^= 1;
      __syncthreads();
    }
  }

  // Epilogue. C layout: VGPR r, lanes 0-15 -> M=r, N=lane; lanes 16-31 -> M=r+8.
  const int colBase = bn + wn * 64;
  const int rowBase = bm + wm * 64 + ((lane >> 4) * 8);
  const int cl = lane & 15;
#pragma unroll
  for (int tm = 0; tm < 4; ++tm) {
#pragma unroll
    for (int tn = 0; tn < 4; ++tn) {
      const int col = colBase + tn * 16 + cl;
      float bv = 0.f;
      if constexpr (EPI == 0 || EPI == 1 || EPI == 4) bv = bias[col];
#pragma unroll
      for (int r = 0; r < 8; ++r) {
        const int row = rowBase + tm * 16 + r;
        const size_t ci = (size_t)row * ldc + col;
        float acc = c[tm][tn][r] + bv;
        if constexpr (EPI == 0) {
          Ch[ci] = (_Float16)acc;
        } else if constexpr (EPI == 1) {
          Cf[ci] = acc + res[ci];
        } else if constexpr (EPI == 2) {
          Ch[ci] = (_Float16)acc;
        } else if constexpr (EPI == 3) {
          float cd = acc + res[ci];
          Cf[ci] = cd;
          Ch[(size_t)row * ldch + col] = (_Float16)cd;
        } else {  // EPI == 4 : gate blend
          float z = 1.f / (1.f + expf(-acc));
          Cf[ci] = z * cand[ci] + (1.f - z) * st[ci];
        }
      }
    }
  }
}

// ---------------------------------------------------------------------------
// Host orchestration
// ---------------------------------------------------------------------------
extern "C" void kernel_launch(void* const* d_in, const int* in_sizes, int n_in,
                              void* d_out, int out_size, void* d_ws, size_t ws_size,
                              hipStream_t stream) {
  (void)in_sizes; (void)n_in; (void)out_size; (void)ws_size;
  const float* x        = (const float*)d_in[0];
  const float* state    = (const float*)d_in[1];
  const float* g1       = (const float*)d_in[2];
  const float* g2       = (const float*)d_in[3];
  const float* in_w     = (const float*)d_in[4];   // [3*DIM, DIM]
  const float* in_b     = (const float*)d_in[5];   // [3*DIM]
  const float* out_w    = (const float*)d_in[6];   // [DIM, DIM]
  const float* out_b    = (const float*)d_in[7];
  const float* w1       = (const float*)d_in[8];   // [HID, DIM]
  const float* w2       = (const float*)d_in[9];
  const float* w3       = (const float*)d_in[10];  // [DIM, HID]
  const float* gate_w   = (const float*)d_in[11];  // [DIM, 2*DIM]
  const float* gate_b   = (const float*)d_in[12];
  float* out            = (float*)d_out;

  uint8_t* base = (uint8_t*)d_ws;
  size_t off = 0;
  auto alloc = [&](size_t bytes) -> void* {
    void* p = base + off;
    off = (off + bytes + 255) & ~(size_t)255;
    return p;
  };
  const size_t BD = (size_t)BATCH * DIM;
  float*    hF   = (float*)alloc(BD * 4);                    // h (pre-norm); reused as candidate
  float*    h2F  = (float*)alloc(BD * 4);                    // h after attention residual
  _Float16* hnH  = (_Float16*)alloc(BD * 2);                 // rmsnorm(h)
  _Float16* vH   = (_Float16*)alloc(BD * 2);                 // v projection; reused as hn2
  _Float16* catH = (_Float16*)alloc((size_t)BATCH * GKDIM * 2);  // [cand | state] f16
  _Float16* t1H  = (_Float16*)alloc((size_t)BATCH * HIDP * 2);
  _Float16* t2H  = (_Float16*)alloc((size_t)BATCH * HIDP * 2);
  _Float16* WvH  = (_Float16*)alloc((size_t)DIM * DIM * 2);
  _Float16* WoH  = (_Float16*)alloc((size_t)DIM * DIM * 2);
  _Float16* W1H  = (_Float16*)alloc((size_t)HIDP * DIM * 2);
  _Float16* W2H  = (_Float16*)alloc((size_t)HIDP * DIM * 2);
  _Float16* W3H  = (_Float16*)alloc((size_t)DIM * HIDP * 2);
  _Float16* WgH  = (_Float16*)alloc((size_t)DIM * GKDIM * 2);
  float*    candF = hF;   // alias: h dead after G2
  _Float16* hn2H  = vH;   // alias: v dead after G2

  auto cgrid = [](long long total) { return dim3((unsigned)((total + 255) / 256)); };

  // Weight conversion (q,k slices of in_proj are dead: softmax over 1 key == 1)
  convert_pad_kernel<<<cgrid((long long)DIM * DIM), 256, 0, stream>>>(
      in_w + (size_t)2 * DIM * DIM, WvH, DIM, DIM, DIM, (long long)DIM * DIM);
  convert_pad_kernel<<<cgrid((long long)DIM * DIM), 256, 0, stream>>>(
      out_w, WoH, DIM, DIM, DIM, (long long)DIM * DIM);
  convert_pad_kernel<<<cgrid((long long)HIDP * DIM), 256, 0, stream>>>(
      w1, W1H, HID, DIM, DIM, (long long)HIDP * DIM);
  convert_pad_kernel<<<cgrid((long long)HIDP * DIM), 256, 0, stream>>>(
      w2, W2H, HID, DIM, DIM, (long long)HIDP * DIM);
  convert_pad_kernel<<<cgrid((long long)DIM * HIDP), 256, 0, stream>>>(
      w3, W3H, DIM, HID, HIDP, (long long)DIM * HIDP);
  convert_pad_kernel<<<cgrid((long long)DIM * GKDIM), 256, 0, stream>>>(
      gate_w, WgH, DIM, GKDIM, GKDIM, (long long)DIM * GKDIM);

  // h = (x+state)*0.7071 ; concat[:,2048:] = f16(state)
  prep_kernel<<<cgrid((long long)BD), 256, 0, stream>>>(x, state, hF, catH);
  // hn = rmsnorm(h, g1)
  rmsnorm_kernel<<<dim3(BATCH), 256, 0, stream>>>(hF, g1, hnH);

  const dim3 blk(256);
  // G1: v = hn @ Wv^T + bv  (f16)
  gemm_kernel<0><<<dim3(DIM / BN, BATCH / BM), blk, 0, stream>>>(
      hnH, WvH, in_b + 2 * DIM, nullptr, nullptr, nullptr,
      nullptr, vH, BATCH, DIM, DIM, DIM, 0);
  // G2: h2 = h + v @ Wo^T + bo  (f32)
  gemm_kernel<1><<<dim3(DIM / BN, BATCH / BM), blk, 0, stream>>>(
      vH, WoH, out_b, hF, nullptr, nullptr,
      h2F, nullptr, BATCH, DIM, DIM, DIM, 0);
  // hn2 = rmsnorm(h2, g2)
  rmsnorm_kernel<<<dim3(BATCH), 256, 0, stream>>>(h2F, g2, hn2H);
  // G3/G4: t1 = hn2 @ w1^T ; t2 = hn2 @ w2^T   (f16, N padded to 5376)
  gemm_kernel<2><<<dim3(HIDP / BN, BATCH / BM), blk, 0, stream>>>(
      hn2H, W1H, nullptr, nullptr, nullptr, nullptr,
      nullptr, t1H, BATCH, HIDP, DIM, HIDP, 0);
  gemm_kernel<2><<<dim3(HIDP / BN, BATCH / BM), blk, 0, stream>>>(
      hn2H, W2H, nullptr, nullptr, nullptr, nullptr,
      nullptr, t2H, BATCH, HIDP, DIM, HIDP, 0);
  // t1 = silu(t1) * t2
  swiglu_kernel<<<cgrid((long long)BATCH * HIDP), 256, 0, stream>>>(
      t1H, t2H, (long long)BATCH * HIDP);
  // G5: cand = h2 + t1 @ w3^T  (f32) and f16 copy into concat[:, :2048]
  gemm_kernel<3><<<dim3(DIM / BN, BATCH / BM), blk, 0, stream>>>(
      t1H, W3H, nullptr, h2F, nullptr, nullptr,
      candF, catH, BATCH, DIM, HIDP, DIM, GKDIM);
  // G6: z = sigmoid(concat @ gate_w^T + gb) ; out = z*cand + (1-z)*state
  gemm_kernel<4><<<dim3(DIM / BN, BATCH / BM), blk, 0, stream>>>(
      catH, WgH, gate_b, nullptr, candF, state,
      out, nullptr, BATCH, DIM, GKDIM, DIM, 0);
}
